// MoELayer_8177617731889
// MI455X (gfx1250) — matrix operand order repaired and checked by reading, not compiled, not run on hardware
//
#include <hip/hip_runtime.h>
#include <hip/hip_bf16.h>

typedef __attribute__((ext_vector_type(16))) __bf16 v16bf;
typedef __attribute__((ext_vector_type(8)))  __bf16 v8bf;
typedef __attribute__((ext_vector_type(8)))  float  v8f;

#define NTOK 2048
#define DDIM 1024
#define FDIM 2048
#define NEXP 8
#define TM 128
#define TN 128
#define KC 32
#define KP 48                // padded LDS K stride in bf16 elems (96B, 16B-aligned)
#define MAX_MT (NTOK / TM)   // 16 M-tiles worst case per expert

// ---------------------------------------------------------------------------
// Router: logits = x @ Wr^T, top-2, renormalized gates, per-expert buckets.
// One wave (32 lanes) per token. bucket entry = (token<<1)|k  ("slot").
// ---------------------------------------------------------------------------
__global__ __launch_bounds__(256) void router_kernel(
    const float* __restrict__ x, const float* __restrict__ Wr,
    float* __restrict__ gate_w, int* __restrict__ bucket, int* __restrict__ counts)
{
  const int token = blockIdx.x * 8 + (threadIdx.x >> 5);
  const int lane  = threadIdx.x & 31;
  if (token >= NTOK) return;

  const float* xr = x + (size_t)token * DDIM;
  float acc[NEXP];
#pragma unroll
  for (int e = 0; e < NEXP; ++e) acc[e] = 0.f;

  for (int d = lane; d < DDIM; d += 32) {
    const float xv = xr[d];
#pragma unroll
    for (int e = 0; e < NEXP; ++e) acc[e] += xv * Wr[e * DDIM + d];
  }
#pragma unroll
  for (int off = 16; off > 0; off >>= 1) {
#pragma unroll
    for (int e = 0; e < NEXP; ++e) acc[e] += __shfl_xor(acc[e], off, 32);
  }

  if (lane == 0) {
    int i0 = 0;
#pragma unroll
    for (int e = 1; e < NEXP; ++e) if (acc[e] > acc[i0]) i0 = e;
    int i1 = -1;
#pragma unroll
    for (int e = 0; e < NEXP; ++e) {
      if (e == i0) continue;
      if (i1 < 0 || acc[e] > acc[i1]) i1 = e;
    }
    // softmax + top-2 renorm == 2-way softmax over the top-2 logits
    const float w0 = 1.f / (1.f + __expf(acc[i1] - acc[i0]));
    gate_w[token * 2 + 0] = w0;
    gate_w[token * 2 + 1] = 1.f - w0;
    const int p0 = atomicAdd(&counts[i0], 1);
    bucket[i0 * NTOK + p0] = (token << 1) | 0;
    const int p1 = atomicAdd(&counts[i1], 1);
    bucket[i1 * NTOK + p1] = (token << 1) | 1;
  }
}

// ---------------------------------------------------------------------------
// Gathered GEMM: C[M,N] = A[M,K] * W[e][N,K]^T  (both K-contiguous)
// fc1: A = x (fp32, row = slot>>1), epilogue relu^2 -> bf16 act[slot][N]
// fc2: A = act (bf16, row = slot), epilogue -> f32 y[slot][N]
// 256 threads = 8 waves; block tile 128x128, wave tile 32x64, KC=32.
// Software-pipelined global->reg prefetch + double-buffered LDS (1 barrier/iter).
// ---------------------------------------------------------------------------
__device__ inline v8bf cvt8(float4 a, float4 b) {
  v8bf v;
  v[0] = (__bf16)a.x; v[1] = (__bf16)a.y; v[2] = (__bf16)a.z; v[3] = (__bf16)a.w;
  v[4] = (__bf16)b.x; v[5] = (__bf16)b.y; v[6] = (__bf16)b.z; v[7] = (__bf16)b.w;
  return v;
}

template <int K_TOT, int N_TOT, bool A_IS_F32, bool RELU_SQ_BF16>
__global__ __launch_bounds__(256) void moe_gemm(
    const void* __restrict__ Abase_v,
    const float* __restrict__ Wall,     // [E][N_TOT][K_TOT]
    void* __restrict__ Out,             // [2*NTOK][N_TOT] (bf16 or f32)
    const int* __restrict__ bucket,     // [E][NTOK], entries = slot
    const int* __restrict__ counts)     // [E]
{
  const int e   = blockIdx.x / MAX_MT;
  const int mt  = blockIdx.x % MAX_MT;
  const int cnt = counts[e];
  const int m0  = mt * TM;
  if (m0 >= cnt) return;
  const int n0 = blockIdx.y * TN;

  __shared__ __align__(16) __bf16 As[2][TM][KP];
  __shared__ __align__(16) __bf16 Bs[2][TN][KP];
  __shared__ __align__(16) int slotS[TM];

  const int t = threadIdx.x;
  if (t < TM) {
    int m = m0 + t;
    if (m >= cnt) m = cnt - 1;                 // clamp; stores are bounds-checked
    slotS[t] = bucket[e * NTOK + m];
  }
  __syncthreads();

  // per-thread staging assignment: 128 rows x 2 threads, 16 elems each
  const int row = t >> 1;
  const int seg = (t & 1) * 16;

  const float* Wrow = Wall + ((size_t)e * N_TOT + (n0 + row)) * (size_t)K_TOT + seg;

  const float*  Af32 = nullptr;
  const __bf16* Abf  = nullptr;
  if constexpr (A_IS_F32) {
    Af32 = (const float*)Abase_v + (size_t)(slotS[row] >> 1) * K_TOT + seg;
  } else {
    Abf  = (const __bf16*)Abase_v + (size_t)slotS[row] * K_TOT + seg;
  }

  const int wid  = t >> 5;
  const int lane = t & 31;
  const int wm   = wid & 3;        // 4 wave-rows  (32 M each)
  const int wn   = wid >> 2;       // 2 wave-cols  (64 N each)
  const int half = lane >> 4;
  const int l16  = lane & 15;

  v8f c[2][4];
#pragma unroll
  for (int i = 0; i < 2; ++i)
#pragma unroll
    for (int j = 0; j < 4; ++j)
      c[i][j] = v8f{0.f, 0.f, 0.f, 0.f, 0.f, 0.f, 0.f, 0.f};

  // -------- register staging for software pipeline --------
  float4 arf[4];   // fc1 A: 16 fp32
  v8bf   arb[2];   // fc2 A: 16 bf16
  float4 brf[4];   // B: 16 fp32 (weights)

#define LOAD_TILE(K0)                                                   \
  do {                                                                  \
    if constexpr (A_IS_F32) {                                           \
      const float* s_ = Af32 + (K0);                                    \
      arf[0] = *(const float4*)(s_);                                    \
      arf[1] = *(const float4*)(s_ + 4);                                \
      arf[2] = *(const float4*)(s_ + 8);                                \
      arf[3] = *(const float4*)(s_ + 12);                               \
    } else {                                                            \
      arb[0] = *(const v8bf*)(Abf + (K0));                              \
      arb[1] = *(const v8bf*)(Abf + (K0) + 8);                          \
    }                                                                   \
    const float* w_ = Wrow + (K0);                                      \
    brf[0] = *(const float4*)(w_);                                      \
    brf[1] = *(const float4*)(w_ + 4);                                  \
    brf[2] = *(const float4*)(w_ + 8);                                  \
    brf[3] = *(const float4*)(w_ + 12);                                 \
  } while (0)

  LOAD_TILE(0);

  int p = 0;
  for (int k0 = 0; k0 < K_TOT; k0 += KC) {
    // ---- commit staged regs to LDS buffer p (with fp32->bf16 convert)
    if constexpr (A_IS_F32) {
      *(v8bf*)&As[p][row][seg]     = cvt8(arf[0], arf[1]);
      *(v8bf*)&As[p][row][seg + 8] = cvt8(arf[2], arf[3]);
    } else {
      *(v8bf*)&As[p][row][seg]     = arb[0];
      *(v8bf*)&As[p][row][seg + 8] = arb[1];
    }
    *(v8bf*)&Bs[p][row][seg]     = cvt8(brf[0], brf[1]);
    *(v8bf*)&Bs[p][row][seg + 8] = cvt8(brf[2], brf[3]);
    __syncthreads();

    // ---- prefetch next K slice into regs (overlaps with WMMA below)
    if (k0 + KC < K_TOT) LOAD_TILE(k0 + KC);

    // ---- fragments per CDNA5 VGPR layouts
    v16bf a[2], b[4];
#pragma unroll
    for (int i = 0; i < 2; ++i) {
      const __bf16* ap = &As[p][wm * 32 + i * 16 + l16][half * 8];
      v8bf lo = *(const v8bf*)ap;           // K = half*8 + 0..7
      v8bf hi = *(const v8bf*)(ap + 16);    // K = half*8 + 16..23
      a[i] = __builtin_shufflevector(lo, hi, 0,1,2,3,4,5,6,7,8,9,10,11,12,13,14,15);
    }
#pragma unroll
    for (int j = 0; j < 4; ++j) {
      const __bf16* bp = &Bs[p][wn * 64 + j * 16 + l16][half * 8];
      v8bf lo = *(const v8bf*)bp;           // K = half*8 + 0..7
      v8bf hi = *(const v8bf*)(bp + 16);    // K = half*8 + 16..23
      b[j] = __builtin_shufflevector(lo, hi, 0,1,2,3,4,5,6,7,8,9,10,11,12,13,14,15);
    }
#pragma unroll
    for (int i = 0; i < 2; ++i)
#pragma unroll
      for (int j = 0; j < 4; ++j)
        c[i][j] = __builtin_amdgcn_wmma_f32_16x16x32_bf16(
            false, a[i], false, b[j], (short)0, c[i][j], false, false);

    p ^= 1;   // ping-pong; single barrier per iter is safe with double buffer
  }
#undef LOAD_TILE

  // --- epilogue: C lane layout M = r + 8*half, N = l16; slots preloaded x16
#pragma unroll
  for (int i = 0; i < 2; ++i) {
    const int mbase = wm * 32 + i * 16 + half * 8;
    const int4 q0 = *(const int4*)&slotS[mbase];
    const int4 q1 = *(const int4*)&slotS[mbase + 4];
    const int slr[8] = {q0.x, q0.y, q0.z, q0.w, q1.x, q1.y, q1.z, q1.w};
#pragma unroll
    for (int r = 0; r < 8; ++r) {
      if (m0 + mbase + r < cnt) {
        const size_t rowoff = (size_t)slr[r] * N_TOT;
#pragma unroll
        for (int j = 0; j < 4; ++j) {
          const int n = n0 + wn * 64 + j * 16 + l16;
          float v = c[i][j][r];
          if constexpr (RELU_SQ_BF16) {
            v = fmaxf(v, 0.f);
            v *= v;
            ((__bf16*)Out)[rowoff + n] = (__bf16)v;
          } else {
            ((float*)Out)[rowoff + n] = v;
          }
        }
      }
    }
  }
}

// ---------------------------------------------------------------------------
// Combine: out[n,:] = w0 * y[2n,:] + w1 * y[2n+1,:]   (deterministic, no atomics)
// ---------------------------------------------------------------------------
__global__ __launch_bounds__(256) void combine_kernel(
    const float* __restrict__ ybuf, const float* __restrict__ gate_w,
    float* __restrict__ out)
{
  const int idx = blockIdx.x * blockDim.x + threadIdx.x;   // over N*D/4
  const int n  = idx >> 8;                                 // D/4 = 256 per token
  const int d4 = idx & 255;
  const float w0 = gate_w[2 * n + 0];
  const float w1 = gate_w[2 * n + 1];
  const float4 y0 = ((const float4*)(ybuf + (size_t)(2 * n)     * DDIM))[d4];
  const float4 y1 = ((const float4*)(ybuf + (size_t)(2 * n + 1) * DDIM))[d4];
  float4 o;
  o.x = w0 * y0.x + w1 * y1.x;
  o.y = w0 * y0.y + w1 * y1.y;
  o.z = w0 * y0.z + w1 * y1.z;
  o.w = w0 * y0.w + w1 * y1.w;
  ((float4*)out)[idx] = o;
}

// ---------------------------------------------------------------------------
extern "C" void kernel_launch(void* const* d_in, const int* in_sizes, int n_in,
                              void* d_out, int out_size, void* d_ws, size_t ws_size,
                              hipStream_t stream) {
  const float* x  = (const float*)d_in[0];
  const float* Wr = (const float*)d_in[1];
  const float* W1 = (const float*)d_in[2];
  const float* W2 = (const float*)d_in[3];
  float* out = (float*)d_out;

  char* ws = (char*)d_ws;
  size_t off = 0;
  auto carve = [&](size_t bytes) -> char* {
    char* p = ws + off;
    off = (off + bytes + 255) & ~(size_t)255;
    return p;
  };
  int*   counts = (int*)  carve(NEXP * sizeof(int));
  float* gate_w = (float*)carve((size_t)NTOK * 2 * sizeof(float));
  int*   bucket = (int*)  carve((size_t)NEXP * NTOK * sizeof(int));
  void*  act    = (void*) carve((size_t)2 * NTOK * FDIM * 2);              // bf16
  float* ybuf   = (float*)carve((size_t)2 * NTOK * DDIM * sizeof(float));  // f32

  hipMemsetAsync(counts, 0, NEXP * sizeof(int), stream);

  router_kernel<<<NTOK / 8, 256, 0, stream>>>(x, Wr, gate_w, bucket, counts);

  dim3 g1(NEXP * MAX_MT, FDIM / TN);   // (128, 16)
  moe_gemm<DDIM, FDIM, true, true><<<g1, 256, 0, stream>>>(
      (const void*)x, W1, act, bucket, counts);

  dim3 g2(NEXP * MAX_MT, DDIM / TN);   // (128, 8)
  moe_gemm<FDIM, DDIM, false, false><<<g2, 256, 0, stream>>>(
      (const void*)act, W2, (void*)ybuf, bucket, counts);

  combine_kernel<<<(NTOK * DDIM / 4) / 256, 256, 0, stream>>>(ybuf, gate_w, out);
}
